// Attention_11055245820093
// MI455X (gfx1250) — matrix operand order repaired and checked by reading, not compiled, not run on hardware
//
#include <hip/hip_runtime.h>
#include <math.h>

#define NB   64
#define NH   12
#define ND   64
#define NC   768
#define NTOK 197
#define NPAD 224      // 14 * 16 = 7 * 32
#define CTL  14       // column tiles in attention
#define M1   12608    // NB * NTOK = 197 * 64
#define NEG_BIG (-3.0e38f)

typedef __attribute__((ext_vector_type(16))) __bf16 v16bf;
typedef __attribute__((ext_vector_type(8)))  float  v8f;

union ABu { v16bf v; unsigned int u[8]; };

__device__ __forceinline__ unsigned short f2bf(float f) {
  unsigned int u = __float_as_uint(f);
  u += 0x7FFFu + ((u >> 16) & 1u);
  return (unsigned short)(u >> 16);
}
__device__ __forceinline__ unsigned int pk2bf(float lo, float hi) {
  return (unsigned int)f2bf(lo) | ((unsigned int)f2bf(hi) << 16);
}
__device__ __forceinline__ v8f wmma_bf16(const ABu& a, const ABu& b, v8f c) {
  return __builtin_amdgcn_wmma_f32_16x16x32_bf16(false, a.v, false, b.v,
                                                 (short)0, c, false, false);
}

// ---------------------------------------------------------------- zero pads
__global__ __launch_bounds__(256) void zero_ws_kernel(uint4* __restrict__ p, int n) {
  int i = blockIdx.x * 256 + threadIdx.x;
  if (i < n) p[i] = make_uint4(0u, 0u, 0u, 0u);
}

// ------------------------------------------------- fp32 -> bf16 (one shot)
__global__ __launch_bounds__(256) void cvt_bf16_kernel(const float2* __restrict__ src,
                                                       unsigned int* __restrict__ dst,
                                                       int n2) {
  int i = blockIdx.x * 256 + threadIdx.x;
  if (i < n2) {
    float2 p = src[i];
    dst[i] = pk2bf(p.x, p.y);
  }
}

// --------------------------------------------- padded relative-position bias
__global__ __launch_bounds__(256) void rpb_kernel(const float* __restrict__ rel_table,
                                                  const int*   __restrict__ rel_index,
                                                  float* __restrict__ rpb) {
  int i = blockIdx.x * 256 + threadIdx.x;
  if (i >= NH * NPAD * NPAD) return;
  int h   = i / (NPAD * NPAD);
  int rm  = i % (NPAD * NPAD);
  int row = rm / NPAD;
  int col = rm % NPAD;
  float v = 0.0f;
  if (row >= 1 && row < NTOK && col >= 1 && col < NTOK) {
    int idx = rel_index[(row - 1) * 196 + (col - 1)];
    v = rel_table[idx * NH + h];
  }
  rpb[i] = v;
}

// -------------------------------------------------------------- QKV GEMM
// 64x64 register-blocked wave tile: A frags reused over 4 N-tiles, B frags
// over 4 M-tiles -> ~1 b128 load per WMMA.  Scatter into Q (scaled),
// K (B,H,224,64) and V transposed (B,H,64,224), all bf16.
__global__ __launch_bounds__(128) void qkv_gemm_kernel(const unsigned short* __restrict__ xb,
                                                       const unsigned short* __restrict__ wb,
                                                       const float* __restrict__ bias,
                                                       unsigned short* __restrict__ Qb,
                                                       unsigned short* __restrict__ Kb,
                                                       unsigned short* __restrict__ Vt) {
  int wid  = blockIdx.x * 4 + (threadIdx.x >> 5);
  int lane = threadIdx.x & 31;
  int mt = wid / 36;           // 197 row strips of 64
  int n0 = (wid % 36) * 64;    // 36 col strips of 64 over 2304
  int l16 = lane & 15;
  int kbA = (lane < 16) ? 0 : 16;
  int kbB = (lane < 16) ? 0 : 16;

  v8f z = {};
  v8f acc[16];
#pragma unroll
  for (int i = 0; i < 16; ++i) acc[i] = z;

  const unsigned int* ap[4];
#pragma unroll
  for (int mi = 0; mi < 4; ++mi)
    ap[mi] = (const unsigned int*)(xb + (size_t)(mt * 64 + mi * 16 + l16) * NC);

  for (int k0 = 0; k0 < NC; k0 += 32) {
    __builtin_prefetch(ap[0] + ((k0 + 32) >> 1), 0, 1);   // global_prefetch_b8
    ABu a[4];
#pragma unroll
    for (int mi = 0; mi < 4; ++mi) {
#pragma unroll
      for (int j = 0; j < 4; ++j) {
        a[mi].u[j]     = ap[mi][(k0 + kbA + 2 * j) >> 1];
        a[mi].u[j + 4] = ap[mi][(k0 + kbA + 8 + 2 * j) >> 1];
      }
    }
    ABu b[4];
#pragma unroll
    for (int t = 0; t < 4; ++t) {
      const uint4* bp = (const uint4*)(wb + (size_t)(n0 + t * 16 + l16) * NC + k0 + kbB);
      uint4 q0 = bp[0], q1 = bp[1];
      b[t].u[0] = q0.x; b[t].u[1] = q0.y; b[t].u[2] = q0.z; b[t].u[3] = q0.w;
      b[t].u[4] = q1.x; b[t].u[5] = q1.y; b[t].u[6] = q1.z; b[t].u[7] = q1.w;
    }
#pragma unroll
    for (int mi = 0; mi < 4; ++mi)
#pragma unroll
      for (int t = 0; t < 4; ++t)
        acc[mi * 4 + t] = wmma_bf16(a[mi], b[t], acc[mi * 4 + t]);
  }

  int rhalf = (lane >> 4) << 3;
#pragma unroll
  for (int mi = 0; mi < 4; ++mi) {
#pragma unroll
    for (int t = 0; t < 4; ++t) {
#pragma unroll
      for (int r = 0; r < 8; ++r) {
        int m  = mt * 64 + mi * 16 + r + rhalf;
        int nc = n0 + t * 16 + l16;
        float val = acc[mi * 4 + t][r] + bias[nc];
        int s   = nc / NC;
        int rem = nc % NC;
        int h = rem >> 6, d = rem & 63;
        int bb = m / NTOK, tok = m % NTOK;
        size_t qki = (((size_t)bb * NH + h) * NPAD + tok) * ND + d;
        if (s == 0)      Qb[qki] = f2bf(val * 0.125f);   // fold scale = D^-0.5
        else if (s == 1) Kb[qki] = f2bf(val);
        else Vt[(((size_t)bb * NH + h) * ND + d) * NPAD + tok] = f2bf(val);
      }
    }
  }
}

// -------------------------------------------------------------- attention
// one wave per (b, h, 16-row query tile):
//   S = Q K^T (WMMA), +bias, /temp, diag mask, softmax, P->LDS(bf16), O = P V (WMMA)
__global__ __launch_bounds__(32) void attn_kernel(const unsigned short* __restrict__ Qb,
                                                  const unsigned short* __restrict__ Kb,
                                                  const unsigned short* __restrict__ Vt,
                                                  const float* __restrict__ rpb,
                                                  const float* __restrict__ log_temp,
                                                  unsigned short* __restrict__ Oc) {
  __shared__ unsigned short Pl[16 * NPAD];

  int bid = blockIdx.x;           // NB*NH*14
  int rt  = bid % CTL;
  int bh  = bid / CTL;
  int h   = bh % NH;
  int bb  = bh / NH;
  int lane = threadIdx.x & 31;
  int l16  = lane & 15;
  int kbA  = (lane < 16) ? 0 : 16;
  int kbB  = (lane < 16) ? 0 : 16;
  int rl   = (lane >> 4) << 3;    // row half offset for C fragments

  const unsigned short* Qp = Qb + (((size_t)bb * NH + h) * NPAD) * ND;
  const unsigned short* Kp = Kb + (((size_t)bb * NH + h) * NPAD) * ND;
  const unsigned short* Vp = Vt + (((size_t)bb * NH + h) * ND) * NPAD;
  const float* rpbh = rpb + (size_t)h * NPAD * NPAD;

  v8f z = {};
  v8f sacc[CTL];
#pragma unroll
  for (int i = 0; i < CTL; ++i) sacc[i] = z;

  // ---- S = Q K^T : two 32-wide k-steps over D=64
  ABu afr[2];
  {
    const unsigned int* qp32 = (const unsigned int*)(Qp + (size_t)(rt * 16 + l16) * ND);
#pragma unroll
    for (int kk = 0; kk < 2; ++kk) {
#pragma unroll
      for (int j = 0; j < 4; ++j) {
        afr[kk].u[j]     = qp32[(kk * 32 + kbA + 2 * j) >> 1];
        afr[kk].u[j + 4] = qp32[(kk * 32 + kbA + 8 + 2 * j) >> 1];
      }
    }
  }
#pragma unroll
  for (int ct = 0; ct < CTL; ++ct) {
    const unsigned short* kp = Kp + (size_t)(ct * 16 + l16) * ND;
#pragma unroll
    for (int kk = 0; kk < 2; ++kk) {
      const uint4* p = (const uint4*)(kp + kk * 32 + kbB);
      uint4 q0 = p[0], q1 = p[1];
      ABu bf;
      bf.u[0] = q0.x; bf.u[1] = q0.y; bf.u[2] = q0.z; bf.u[3] = q0.w;
      bf.u[4] = q1.x; bf.u[5] = q1.y; bf.u[6] = q1.z; bf.u[7] = q1.w;
      sacc[ct] = wmma_bf16(afr[kk], bf, sacc[ct]);
    }
  }

  // ---- bias, temperature, mask, softmax (rows live in 16-lane halves)
  float temp = logf(1.0f + expf(log_temp[h]));   // softplus
  float rinv = 1.0f / temp;
#pragma unroll
  for (int r = 0; r < 8; ++r) {
    int row = rt * 16 + r + rl;
    float mx = NEG_BIG;
#pragma unroll
    for (int ct = 0; ct < CTL; ++ct) {
      int col = ct * 16 + l16;
      float v = (sacc[ct][r] + rpbh[row * NPAD + col]) * rinv;
      if (col >= NTOK || (row == col && row >= 1)) v = NEG_BIG;
      sacc[ct][r] = v;
      mx = fmaxf(mx, v);
    }
#pragma unroll
    for (int m = 1; m < 16; m <<= 1) mx = fmaxf(mx, __shfl_xor(mx, m, 32));
    float sum = 0.0f;
#pragma unroll
    for (int ct = 0; ct < CTL; ++ct) {
      float p = __expf(sacc[ct][r] - mx);
      sacc[ct][r] = p;
      sum += p;
    }
#pragma unroll
    for (int m = 1; m < 16; m <<= 1) sum += __shfl_xor(sum, m, 32);
    float inv = 1.0f / sum;
#pragma unroll
    for (int ct = 0; ct < CTL; ++ct) {
      int col = ct * 16 + l16;
      Pl[(r + rl) * NPAD + col] = f2bf(sacc[ct][r] * inv);
    }
  }
  __syncthreads();

  // ---- O = P V : P from LDS in A-fragment layout, V^T gives contiguous B frags
  v8f oacc[4] = { z, z, z, z };
  const unsigned int* pl32 = (const unsigned int*)(&Pl[l16 * NPAD]);
  for (int kt = 0; kt < 7; ++kt) {
    int k0 = kt * 32;
    ABu a;
#pragma unroll
    for (int j = 0; j < 4; ++j) {
      a.u[j]     = pl32[(k0 + kbA + 2 * j) >> 1];
      a.u[j + 4] = pl32[(k0 + kbA + 8 + 2 * j) >> 1];
    }
#pragma unroll
    for (int t = 0; t < 4; ++t) {
      const uint4* vp = (const uint4*)(Vp + (size_t)(t * 16 + l16) * NPAD + k0 + kbB);
      uint4 q0 = vp[0], q1 = vp[1];
      ABu bf;
      bf.u[0] = q0.x; bf.u[1] = q0.y; bf.u[2] = q0.z; bf.u[3] = q0.w;
      bf.u[4] = q1.x; bf.u[5] = q1.y; bf.u[6] = q1.z; bf.u[7] = q1.w;
      oacc[t] = wmma_bf16(a, bf, oacc[t]);
    }
  }

#pragma unroll
  for (int t = 0; t < 4; ++t) {
#pragma unroll
    for (int r = 0; r < 8; ++r) {
      int tok = rt * 16 + r + rl;
      if (tok < NTOK) {
        int d = t * 16 + l16;
        Oc[((size_t)bb * NTOK + tok) * NC + h * ND + d] = f2bf(oacc[t][r]);
      }
    }
  }
}

// -------------------------------------------------------------- projection
// 64x64 register-blocked wave tile; fp32 output with bias
__global__ __launch_bounds__(128) void proj_gemm_kernel(const unsigned short* __restrict__ Oc,
                                                        const unsigned short* __restrict__ wb,
                                                        const float* __restrict__ bias,
                                                        float* __restrict__ out) {
  int wid  = blockIdx.x * 4 + (threadIdx.x >> 5);
  int lane = threadIdx.x & 31;
  int mt = wid / 12;           // 197 row strips of 64
  int n0 = (wid % 12) * 64;    // 12 col strips of 64 over 768
  int l16 = lane & 15;
  int kbA = (lane < 16) ? 0 : 16;
  int kbB = (lane < 16) ? 0 : 16;

  v8f z = {};
  v8f acc[16];
#pragma unroll
  for (int i = 0; i < 16; ++i) acc[i] = z;

  const unsigned int* ap[4];
#pragma unroll
  for (int mi = 0; mi < 4; ++mi)
    ap[mi] = (const unsigned int*)(Oc + (size_t)(mt * 64 + mi * 16 + l16) * NC);

  for (int k0 = 0; k0 < NC; k0 += 32) {
    __builtin_prefetch(ap[0] + ((k0 + 32) >> 1), 0, 1);   // global_prefetch_b8
    ABu a[4];
#pragma unroll
    for (int mi = 0; mi < 4; ++mi) {
#pragma unroll
      for (int j = 0; j < 4; ++j) {
        a[mi].u[j]     = ap[mi][(k0 + kbA + 2 * j) >> 1];
        a[mi].u[j + 4] = ap[mi][(k0 + kbA + 8 + 2 * j) >> 1];
      }
    }
    ABu b[4];
#pragma unroll
    for (int t = 0; t < 4; ++t) {
      const uint4* bp = (const uint4*)(wb + (size_t)(n0 + t * 16 + l16) * NC + k0 + kbB);
      uint4 q0 = bp[0], q1 = bp[1];
      b[t].u[0] = q0.x; b[t].u[1] = q0.y; b[t].u[2] = q0.z; b[t].u[3] = q0.w;
      b[t].u[4] = q1.x; b[t].u[5] = q1.y; b[t].u[6] = q1.z; b[t].u[7] = q1.w;
    }
#pragma unroll
    for (int mi = 0; mi < 4; ++mi)
#pragma unroll
      for (int t = 0; t < 4; ++t)
        acc[mi * 4 + t] = wmma_bf16(a[mi], b[t], acc[mi * 4 + t]);
  }

  int rhalf = (lane >> 4) << 3;
#pragma unroll
  for (int mi = 0; mi < 4; ++mi) {
#pragma unroll
    for (int t = 0; t < 4; ++t) {
#pragma unroll
      for (int r = 0; r < 8; ++r) {
        int m  = mt * 64 + mi * 16 + r + rhalf;
        int nc = n0 + t * 16 + l16;
        out[(size_t)m * NC + nc] = acc[mi * 4 + t][r] + bias[nc];
      }
    }
  }
}

// ---------------------------------------------------------------- launcher
extern "C" void kernel_launch(void* const* d_in, const int* in_sizes, int n_in,
                              void* d_out, int out_size, void* d_ws, size_t ws_size,
                              hipStream_t stream) {
  (void)in_sizes; (void)n_in; (void)out_size; (void)ws_size;
  const float* x         = (const float*)d_in[0];
  const float* qkv_w     = (const float*)d_in[1];
  const float* qkv_b     = (const float*)d_in[2];
  const float* proj_w    = (const float*)d_in[3];
  const float* proj_b    = (const float*)d_in[4];
  const float* rel_table = (const float*)d_in[5];
  const float* log_temp  = (const float*)d_in[6];
  const int*   rel_index = (const int*)d_in[7];

  char* ws = (char*)d_ws;
  const size_t QK_BYTES  = (size_t)NB * NH * NPAD * ND * 2;   // 22,020,096
  const size_t RPB_BYTES = (size_t)NH * NPAD * NPAD * 4;      //  2,408,448
  const size_t OC_BYTES  = (size_t)M1 * NC * 2;               // 19,365,888
  const size_t XB_BYTES  = (size_t)M1 * NC * 2;               // 19,365,888
  const size_t WQ_BYTES  = (size_t)3 * NC * NC * 2;           //  3,538,944

  unsigned short* Qb  = (unsigned short*)(ws);
  unsigned short* Kb  = (unsigned short*)(ws + QK_BYTES);
  unsigned short* Vt  = (unsigned short*)(ws + 2 * QK_BYTES);
  float*          rp  = (float*)(ws + 3 * QK_BYTES);
  unsigned short* Oc  = (unsigned short*)(ws + 3 * QK_BYTES + RPB_BYTES);
  unsigned short* xb  = (unsigned short*)(ws + 3 * QK_BYTES + RPB_BYTES + OC_BYTES);
  unsigned short* wqb = (unsigned short*)(ws + 3 * QK_BYTES + RPB_BYTES + OC_BYTES + XB_BYTES);
  unsigned short* wpb = (unsigned short*)(ws + 3 * QK_BYTES + RPB_BYTES + OC_BYTES + XB_BYTES + WQ_BYTES);
  float* out = (float*)d_out;

  // zero Q/K/Vt (covers the pad rows/cols the GEMM never writes)
  int nvec4 = (int)(3 * QK_BYTES / 16);                  // 4,128,768
  zero_ws_kernel<<<(nvec4 + 255) / 256, 256, 0, stream>>>((uint4*)ws, nvec4);

  // one-shot fp32 -> bf16 conversion of x and both weight matrices
  int nx2 = M1 * NC / 2;                                 // 4,841,472
  cvt_bf16_kernel<<<(nx2 + 255) / 256, 256, 0, stream>>>((const float2*)x,
                                                         (unsigned int*)xb, nx2);
  int nwq2 = 3 * NC * NC / 2;                            // 884,736
  cvt_bf16_kernel<<<(nwq2 + 255) / 256, 256, 0, stream>>>((const float2*)qkv_w,
                                                          (unsigned int*)wqb, nwq2);
  int nwp2 = NC * NC / 2;                                // 294,912
  cvt_bf16_kernel<<<(nwp2 + 255) / 256, 256, 0, stream>>>((const float2*)proj_w,
                                                          (unsigned int*)wpb, nwp2);

  // padded relative position bias (H, 224, 224)
  int nrpb = NH * NPAD * NPAD;
  rpb_kernel<<<(nrpb + 255) / 256, 256, 0, stream>>>(rel_table, rel_index, rp);

  // QKV projection: 197 row strips * 36 col strips = 7092 waves / 4 per block
  qkv_gemm_kernel<<<7092 / 4, 128, 0, stream>>>(xb, wqb, qkv_b, Qb, Kb, Vt);

  // attention: one wave per (b, h, 16-row tile)
  attn_kernel<<<NB * NH * CTL, 32, 0, stream>>>(Qb, Kb, Vt, rp, log_temp, Oc);

  // output projection: 197 * 12 = 2364 waves / 4 per block
  proj_gemm_kernel<<<2364 / 4, 128, 0, stream>>>(Oc, wpb, proj_b, out);
}
